// TwoMLPHead_67061619360308
// MI455X (gfx1250) — matrix operand order repaired and checked by reading, not compiled, not run on hardware
//
#include <hip/hip_runtime.h>

// ---------------------------------------------------------------------------
// Relation-Network head for MI455X (gfx1250, wave32, WMMA).
// One batched WMMA GEMM kernel:
//     C = alpha * A(MxK) @ B(NxK)^T  [+ bias] [+ C] [relu]
// f16 WMMA operands with f32 accumulation (v_wmma_f32_16x16x32_f16).
// f32 sources are converted to f16 *inside* the staging path (templated
// operand modes), so no separate conversion passes touch HBM.
// LDS double buffering, register prefetch, unguarded (clamped) b128 staging.
// ---------------------------------------------------------------------------

typedef _Float16 f16;
typedef __attribute__((ext_vector_type(16))) _Float16 v16h;
typedef __attribute__((ext_vector_type(8)))  float    v8f;
typedef __attribute__((ext_vector_type(4)))  float    f32x4;
typedef __attribute__((ext_vector_type(4)))  unsigned u32x4;

#define N_ROIS 2000
#define IN_CH  12544
#define REP    1024
#define NONGT  300
#define EMB    64
#define GROUP  16
#define DG     64          // REP / GROUP
#define KPAD   320         // NONGT padded up to a multiple of 32

#define LSTRIDE 40         // LDS tile row stride in halves (80 B, 16B-aligned rows)
#define LTILE   (64 * LSTRIDE)

// Operand staging modes
#define OP_F16      0      // source already f16
#define OP_F32      1      // source f32, convert while staging
#define OP_F32_RELU 2      // source f32, relu + convert while staging

// Load one 8-element chunk (16B of f16 after conversion) at element offset.
template <int MODE>
__device__ __forceinline__ u32x4 load_chunk(const void* __restrict__ base, long off)
{
    if (MODE == OP_F16) {
        return *reinterpret_cast<const u32x4*>(reinterpret_cast<const f16*>(base) + off);
    } else {
        const float* p = reinterpret_cast<const float*>(base) + off;
        f32x4 x0 = *reinterpret_cast<const f32x4*>(p);
        f32x4 x1 = *reinterpret_cast<const f32x4*>(p + 4);
        if (MODE == OP_F32_RELU) {
            #pragma unroll
            for (int j = 0; j < 4; ++j) {
                x0[j] = fmaxf(x0[j], 0.f);
                x1[j] = fmaxf(x1[j], 0.f);
            }
        }
        union { u32x4 u; f16 h[8]; } t;
        #pragma unroll
        for (int j = 0; j < 4; ++j) {
            t.h[j]     = (f16)x0[j];
            t.h[4 + j] = (f16)x1[j];
        }
        return t.u;
    }
}

// ---------------------------------------------------------------------------
// Generic batched WMMA GEMM.  A: MxK row-major (lda), B: NxK row-major (ldb),
// C: MxN row-major (ldc).  Per-batch element strides sA/sB/sC/sBias via
// blockIdx.z.  Requirements (all call sites satisfy them):
//   - K % 32 == 0
//   - lda, ldb and all batch strides are multiples of 8 elements
//   - A/B base pointers 16B aligned
// OOB staging rows are CLAMPED, not zeroed: garbage only reaches C rows/cols
// that are never stored.  K-direction zero padding is the caller's job.
// ---------------------------------------------------------------------------
template <int AM, int BM>
__global__ __launch_bounds__(256)
void wmma_gemm_kernel(const void* __restrict__ A, long lda, long sA,
                      const void* __restrict__ B, long ldb, long sB,
                      float* Cf, f16* Ch, long ldc, long sC,
                      const float* __restrict__ bias, long sBias,
                      int M, int N, int K, float alpha, int accum, int relu)
{
    alignas(16) __shared__ f16 As[2 * LTILE];   // double-buffered 64x32 tiles
    alignas(16) __shared__ f16 Bs[2 * LTILE];

    const int z = blockIdx.z;
    if (Cf)   Cf   += (long)z * sC;
    if (Ch)   Ch   += (long)z * sC;
    if (bias) bias += (long)z * sBias;

    const int bm0  = blockIdx.y * 64;
    const int bn0  = blockIdx.x * 64;
    const int tid  = threadIdx.x;
    const int lane = tid & 31;
    const int wave = tid >> 5;
    const int waveM = wave & 3;   // 4 waves along M (16 rows each)
    const int waveN = wave >> 2;  // 2 waves along N (32 cols each)

    v8f acc0 = {0.f,0.f,0.f,0.f,0.f,0.f,0.f,0.f};
    v8f acc1 = {0.f,0.f,0.f,0.f,0.f,0.f,0.f,0.f};

    // Staging: each thread owns one 8-element chunk of each 64x32 tile.
    const int srow = tid >> 2;           // 0..63
    const int scol = (tid & 3) * 8;      // 0,8,16,24
    // Clamp rows: OOB rows only feed never-stored C entries.
    const int garow = (bm0 + srow < M) ? (bm0 + srow) : (M - 1);
    const int gbrow = (bn0 + srow < N) ? (bn0 + srow) : (N - 1);
    const long aoff = (long)z * sA + (long)garow * lda + scol;
    const long boff = (long)z * sB + (long)gbrow * ldb + scol;

    // ISA 16-bit A 16x32 layout: lanes 0-15 hold K {0..7,16..23} of row M=lane,
    // lanes 16-31 hold K {8..15,24..31}; consecutive element pairs = consec K.
    const int am    = waveM * 16 + (lane & 15);
    const int koffA = (lane < 16) ? 0 : 8;
    // B 32x16 layout: lanes 0-15 -> N=lane, K {0..7,16..23}; lanes 16-31 -> K {8..15,24..31}.
    const int bnL0  = waveN * 32 + (lane & 15);
    const int bnL1  = bnL0 + 16;
    const int koffB = (lane < 16) ? 0 : 8;

    const int nIt = K >> 5;

    // Prologue: prefetch tile 0 into registers.
    u32x4 aReg = load_chunk<AM>(A, aoff);
    u32x4 bReg = load_chunk<BM>(B, boff);

    for (int it = 0; it < nIt; ++it) {
        f16* Asb = As + (it & 1) * LTILE;
        f16* Bsb = Bs + (it & 1) * LTILE;

        *reinterpret_cast<u32x4*>(&Asb[srow * LSTRIDE + scol]) = aReg;
        *reinterpret_cast<u32x4*>(&Bsb[srow * LSTRIDE + scol]) = bReg;
        __syncthreads();

        // Prefetch next tile while this one is consumed.
        if (it + 1 < nIt) {
            long ko = (long)(it + 1) << 5;
            aReg = load_chunk<AM>(A, aoff + ko);
            bReg = load_chunk<BM>(B, boff + ko);
        }

        union { v16h h; unsigned u[8]; } fa, fb0, fb1;
        #pragma unroll
        for (int p = 0; p < 8; ++p) {
            int ka = (p < 4) ? (koffA + 2 * p) : (16 + koffA + 2 * (p - 4));
            fa.u[p]  = *reinterpret_cast<const unsigned*>(&Asb[am   * LSTRIDE + ka]);
            int kb = (p < 4) ? (koffB + 2 * p) : (16 + koffB + 2 * (p - 4));
            fb0.u[p] = *reinterpret_cast<const unsigned*>(&Bsb[bnL0 * LSTRIDE + kb]);
            fb1.u[p] = *reinterpret_cast<const unsigned*>(&Bsb[bnL1 * LSTRIDE + kb]);
        }
        acc0 = __builtin_amdgcn_wmma_f32_16x16x32_f16(false, fa.h, false, fb0.h,
                                                      (short)0, acc0, false, false);
        acc1 = __builtin_amdgcn_wmma_f32_16x16x32_f16(false, fa.h, false, fb1.h,
                                                      (short)0, acc1, false, false);
        // No trailing barrier: next iteration writes the other LDS buffer; the
        // single barrier per step is sufficient (ping-pong parity).
    }

    // C layout: lanes 0-15 -> N=lane, M=r; lanes 16-31 -> N=lane-16, M=8+r.
    const int mbase = bm0 + waveM * 16 + ((lane < 16) ? 0 : 8);
    const int gn0   = bn0 + waveN * 32 + (lane & 15);
    const int gn1   = gn0 + 16;
    #pragma unroll
    for (int r = 0; r < 8; ++r) {
        int gm = mbase + r;
        if (gm >= M) continue;
        if (gn0 < N) {
            float v = alpha * acc0[r];
            if (bias)  v += bias[gn0];
            if (accum) v += Cf[(long)gm * ldc + gn0];
            if (relu)  v = fmaxf(v, 0.f);
            if (Ch) Ch[(long)gm * ldc + gn0] = (f16)v;
            else    Cf[(long)gm * ldc + gn0] = v;
        }
        if (gn1 < N) {
            float v = alpha * acc1[r];
            if (bias)  v += bias[gn1];
            if (accum) v += Cf[(long)gm * ldc + gn1];
            if (relu)  v = fmaxf(v, 0.f);
            if (Ch) Ch[(long)gm * ldc + gn1] = (f16)v;
            else    Cf[(long)gm * ldc + gn1] = v;
        }
    }
}

// ---------------------------------------------------------------------------
// Position embedding fused with the wg projection + relu + log:
// laff[n,g,m] = log(max(relu(emb(n,m) . wg[g] + bg[g]), 1e-6))
// ---------------------------------------------------------------------------
__global__ __launch_bounds__(256)
void posemb_laff_kernel(float* __restrict__ laff, const float* __restrict__ rois,
                        const float* __restrict__ wg, const float* __restrict__ bg)
{
    __shared__ float wgs[GROUP * EMB];
    __shared__ float bgs[GROUP];
    int tid = threadIdx.x;
    for (int i = tid; i < GROUP * EMB; i += 256) wgs[i] = wg[i];
    if (tid < GROUP) bgs[tid] = bg[tid];
    __syncthreads();

    long idx = (long)blockIdx.x * 256 + tid;
    if (idx >= (long)N_ROIS * NONGT) return;
    int m = (int)(idx % NONGT);
    int n = (int)(idx / NONGT);

    float xn0 = rois[n * 5 + 1], yn0 = rois[n * 5 + 2];
    float xn1 = rois[n * 5 + 3], yn1 = rois[n * 5 + 4];
    float xm0 = rois[m * 5 + 1], ym0 = rois[m * 5 + 2];
    float xm1 = rois[m * 5 + 3], ym1 = rois[m * 5 + 4];
    float wn = xn1 - xn0 + 1.f, hn = yn1 - yn0 + 1.f;
    float wm = xm1 - xm0 + 1.f, hm = ym1 - ym0 + 1.f;
    float cxn = 0.5f * (xn0 + xn1), cyn = 0.5f * (yn0 + yn1);
    float cxm = 0.5f * (xm0 + xm1), cym = 0.5f * (ym0 + ym1);

    float pos[4];
    pos[0] = __logf(fmaxf(fabsf((cxn - cxm) / wn), 0.001f));
    pos[1] = __logf(fmaxf(fabsf((cyn - cym) / hn), 0.001f));
    pos[2] = __logf(wn / wm);
    pos[3] = __logf(hn / hm);

    float acc[GROUP];
    #pragma unroll
    for (int f = 0; f < GROUP; ++f) acc[f] = bgs[f];

    #pragma unroll
    for (int d = 0; d < 4; ++d) {
        #pragma unroll
        for (int j = 0; j < 8; ++j) {
            float t = 100.f * pos[d] * __powf(1000.f, -(float)j * 0.125f);
            float s = __sinf(t), c = __cosf(t);
            int es = d * 16 + j, ec = es + 8;
            #pragma unroll
            for (int f = 0; f < GROUP; ++f)
                acc[f] += s * wgs[f * EMB + es] + c * wgs[f * EMB + ec];
        }
    }
    #pragma unroll
    for (int f = 0; f < GROUP; ++f)
        laff[((long)n * GROUP + f) * NONGT + m] = __logf(fmaxf(acc[f], 1e-6f));
}

// ---------------------------------------------------------------------------
// Row softmax over m (300), one wave32 per (n,g) row; emits f16 rows padded
// with zeros up to KPAD=320 so downstream WMMA K stays a multiple of 32.
// ---------------------------------------------------------------------------
__global__ __launch_bounds__(256)
void softmax_kernel(f16* __restrict__ smh, const float* __restrict__ laff)
{
    const int row  = blockIdx.x * 8 + (threadIdx.x >> 5);
    const int lane = threadIdx.x & 31;
    if (row >= N_ROIS * GROUP) return;

    const float* src = laff + (long)row * NONGT;
    float vals[10];
    float mx = -3.4e38f;
    #pragma unroll
    for (int i = 0; i < 10; ++i) {
        int m = lane + i * 32;
        float v = (m < NONGT) ? src[m] : -3.4e38f;
        vals[i] = v;
        mx = fmaxf(mx, v);
    }
    #pragma unroll
    for (int off = 16; off >= 1; off >>= 1)
        mx = fmaxf(mx, __shfl_xor(mx, off, 32));

    float s = 0.f;
    #pragma unroll
    for (int i = 0; i < 10; ++i) {
        int m = lane + i * 32;
        float e = (m < NONGT) ? __expf(vals[i] - mx) : 0.f;
        vals[i] = e;
        s += e;
    }
    #pragma unroll
    for (int off = 16; off >= 1; off >>= 1)
        s += __shfl_xor(s, off, 32);
    float inv = 1.f / s;

    f16* dst = smh + (long)row * KPAD;
    #pragma unroll
    for (int i = 0; i < 10; ++i) {
        int m = lane + i * 32;
        if (m < NONGT) dst[m] = (f16)(vals[i] * inv);
    }
    if (lane < KPAD - NONGT) dst[NONGT + lane] = (f16)0.f;
}

// ---------------------------------------------------------------------------
// Small helpers.
// ---------------------------------------------------------------------------
__global__ void fill0_f16_kernel(f16* __restrict__ dst, long count)
{
    long i = (long)blockIdx.x * blockDim.x + threadIdx.x;
    if (i < count) dst[i] = (f16)0.f;
}

__global__ void relu_out_kernel(float* __restrict__ dst, const float* __restrict__ src,
                                long count)
{
    long i = (long)blockIdx.x * blockDim.x + threadIdx.x;
    if (i < count) dst[i] = fmaxf(src[i], 0.f);
}

// ---------------------------------------------------------------------------
// Host side
// ---------------------------------------------------------------------------
static inline void gemm(hipStream_t s, int amode, int bmode,
                        const void* A, long lda, long sA,
                        const void* B, long ldb, long sB,
                        float* Cf, f16* Ch, long ldc, long sC,
                        const float* bias, long sBias,
                        int M, int N, int K, float alpha, int accum, int relu, int batch)
{
    dim3 g((N + 63) / 64, (M + 63) / 64, batch);
    if (amode == OP_F16 && bmode == OP_F16)
        wmma_gemm_kernel<OP_F16, OP_F16><<<g, 256, 0, s>>>(A, lda, sA, B, ldb, sB,
            Cf, Ch, ldc, sC, bias, sBias, M, N, K, alpha, accum, relu);
    else if (amode == OP_F32_RELU)
        wmma_gemm_kernel<OP_F32_RELU, OP_F32><<<g, 256, 0, s>>>(A, lda, sA, B, ldb, sB,
            Cf, Ch, ldc, sC, bias, sBias, M, N, K, alpha, accum, relu);
    else
        wmma_gemm_kernel<OP_F32, OP_F32><<<g, 256, 0, s>>>(A, lda, sA, B, ldb, sB,
            Cf, Ch, ldc, sC, bias, sBias, M, N, K, alpha, accum, relu);
}

struct Rel {
    const float *wg, *bg, *wq, *bq, *wk, *bk, *wo, *bo;
};

// feat F (f32, 2000x1024) gets F += relation(F) in place.
// Reassociation: out[n,g,o] = sum_m sm[n,g,m] * P[g][m,o],
// P[g] = feat[:300] . wo[g]^T  (computed transposed as Pt[g][o][m], m-padded).
static void relation(hipStream_t s, float* F, f16* qh, f16* kfh,
                     float* laff, f16* smh, f16* Pt, const float* rois, const Rel& r)
{
    long c;

    // q = feat @ wq^T + bq  -> f16 [2000 x 1024]   (f32 sources, fused cvt)
    gemm(s, OP_F32, OP_F32, F, REP, 0, r.wq, REP, 0, nullptr, qh, REP, 0, r.bq, 0,
         N_ROIS, REP, REP, 1.f, 0, 0, 1);

    // kf = feat[:300] @ wk^T + bk -> f16 [300 x 1024]
    gemm(s, OP_F32, OP_F32, F, REP, 0, r.wk, REP, 0, nullptr, kfh, REP, 0, r.bk, 0,
         NONGT, REP, REP, 1.f, 0, 0, 1);

    // laff[n,g,m] = log(max(aff_w,1e-6))
    c = (long)N_ROIS * NONGT;
    posemb_laff_kernel<<<(unsigned)((c + 255) / 256), 256, 0, s>>>(laff, rois, r.wg, r.bg);

    // laff += (1/sqrt(64)) * q_g @ kf_g^T   (batched over g, accumulate)
    gemm(s, OP_F16, OP_F16, qh, REP, DG, kfh, REP, DG, laff, nullptr,
         (long)GROUP * NONGT, NONGT, nullptr, 0,
         N_ROIS, NONGT, DG, 0.125f, 1, 0, GROUP);

    // softmax over m, write f16 zero-padded to KPAD
    softmax_kernel<<<(N_ROIS * GROUP + 7) / 8, 256, 0, s>>>(smh, laff);

    // Pt[g][o][m] = wo[g] @ feat[:300]^T   (f16, m-padded to 320 via pre-zero)
    c = (long)GROUP * DG * KPAD;
    fill0_f16_kernel<<<(unsigned)((c + 255) / 256), 256, 0, s>>>(Pt, c);
    gemm(s, OP_F32, OP_F32, r.wo, REP, (long)DG * REP, F, REP, 0,
         nullptr, Pt, KPAD, (long)DG * KPAD, nullptr, 0,
         DG, NONGT, REP, 1.f, 0, 0, GROUP);

    // F[:, g*64 + o] += sm_g @ Pt_g^T + bo[g*64 + o]   (residual, batched)
    gemm(s, OP_F16, OP_F16, smh, (long)GROUP * KPAD, KPAD, Pt, KPAD, (long)DG * KPAD,
         F, nullptr, REP, DG, r.bo, DG, N_ROIS, DG, KPAD, 1.f, 1, 0, GROUP);
}

extern "C" void kernel_launch(void* const* d_in, const int* in_sizes, int n_in,
                              void* d_out, int out_size, void* d_ws, size_t ws_size,
                              hipStream_t stream)
{
    (void)in_sizes; (void)n_in; (void)out_size; (void)ws_size;

    const float* x     = (const float*)d_in[0];
    const float* rois  = (const float*)d_in[1];
    const float* fc6_w = (const float*)d_in[2];
    const float* fc6_b = (const float*)d_in[3];
    const float* fc7_w = (const float*)d_in[4];
    const float* fc7_b = (const float*)d_in[5];
    Rel r1 = { (const float*)d_in[6],  (const float*)d_in[7],  (const float*)d_in[8],
               (const float*)d_in[9],  (const float*)d_in[10], (const float*)d_in[11],
               (const float*)d_in[12], (const float*)d_in[13] };
    Rel r2 = { (const float*)d_in[14], (const float*)d_in[15], (const float*)d_in[16],
               (const float*)d_in[17], (const float*)d_in[18], (const float*)d_in[19],
               (const float*)d_in[20], (const float*)d_in[21] };

    // Workspace carve (~81 MB total)
    char* p = (char*)d_ws;
    auto carve = [&](size_t bytes) -> char* {
        char* q = p;
        p += (bytes + 255) & ~(size_t)255;
        return q;
    };
    float* Fa   = (float*)carve((size_t)N_ROIS * REP * 4);            //  8.2 MB (fc6 + rel1)
    float* Fb   = (float*)carve((size_t)N_ROIS * REP * 4);            //  8.2 MB (fc7 + rel2)
    f16*   qh   = (f16*)  carve((size_t)N_ROIS * REP * 2);            //  4.1 MB
    f16*   kfh  = (f16*)  carve((size_t)NONGT * REP * 2);             //  0.6 MB
    float* laff = (float*)carve((size_t)N_ROIS * GROUP * NONGT * 4);  // 38.4 MB
    f16*   smh  = (f16*)  carve((size_t)N_ROIS * GROUP * KPAD * 2);   // 20.5 MB
    f16*   Pt   = (f16*)  carve((size_t)GROUP * DG * KPAD * 2);       //  0.7 MB

    // fc6 = x @ fc6_w^T + fc6_b    (both operands f32, converted in staging)
    gemm(stream, OP_F32, OP_F32, x, IN_CH, 0, fc6_w, IN_CH, 0,
         Fa, nullptr, REP, 0, fc6_b, 0, N_ROIS, REP, IN_CH, 1.f, 0, 0, 1);

    // fc6 += relation1(fc6)
    relation(stream, Fa, qh, kfh, laff, smh, Pt, rois, r1);

    // fc7 = relu(fc6) @ fc7_w^T + fc7_b   (relu fused into A staging)
    gemm(stream, OP_F32_RELU, OP_F32, Fa, REP, 0, fc7_w, REP, 0,
         Fb, nullptr, REP, 0, fc7_b, 0, N_ROIS, REP, REP, 1.f, 0, 0, 1);

    // fc7 += relation2(fc7)
    relation(stream, Fb, qh, kfh, laff, smh, Pt, rois, r2);

    // out = relu(fc7)
    long c = (long)N_ROIS * REP;
    relu_out_kernel<<<(unsigned)((c + 255) / 256), 256, 0, stream>>>((float*)d_out, Fb, c);
}
